// XbmTripletLoss_1769526526574
// MI455X (gfx1250) — matrix operand order repaired and checked by reading, not compiled, not run on hardware
//
#include <hip/hip_runtime.h>
#include <hip/hip_bf16.h>
#include <stdint.h>

// ---------------------------------------------------------------------------
// XBM triplet loss on gfx1250 (MI455X): bf16 WMMA GEMM, two-sweep reduction.
// sim = A(512x512) * B(65536x512)^T, K=512, via v_wmma_f32_16x16x32_bf16.
// B pre-converted to bf16 (L2-resident, 64MB), A pre-packed into WMMA
// fragment layout, B tiles staged with global_load_async_to_lds_b128.
// ---------------------------------------------------------------------------

typedef __attribute__((ext_vector_type(16))) __bf16 v16bf;
typedef __attribute__((ext_vector_type(8)))  float  v8f;

#define R_TOT   512
#define C_TOT   65536
#define K_TOT   512
#define NKS     (K_TOT / 32)
#define MARGIN_F 0.1f
#define NEGBASE_F 0.6f
#define ENC_NEG_INF 0x007FFFFFu   // order-encoding of -inf
#define LDS_BUF_BYTES 10240u      // one B-tile buffer (128 cols x 80B)

__device__ __forceinline__ unsigned encf(float f) {
  unsigned b = __float_as_uint(f);
  return (b & 0x80000000u) ? ~b : (b | 0x80000000u);
}
__device__ __forceinline__ float decf(unsigned u) {
  unsigned b = (u & 0x80000000u) ? (u ^ 0x80000000u) : ~u;
  return __uint_as_float(b);
}
// pack two f32 -> two bf16 (round to nearest even)
__device__ __forceinline__ unsigned pk_bf16(float lo, float hi) {
  unsigned a = __float_as_uint(lo), b = __float_as_uint(hi);
  a += 0x7fffu + ((a >> 16) & 1u);
  b += 0x7fffu + ((b >> 16) & 1u);
  return (a >> 16) | (b & 0xffff0000u);
}

// CDNA5 async copy: global -> LDS, 2 x b128 (one 32B lane chunk), ASYNCcnt.
__device__ __forceinline__ void asyncPairB128(unsigned ldsDst, unsigned gOff,
                                              unsigned long long sBase) {
  asm volatile("global_load_async_to_lds_b128 %0, %1, %2 offset:0\n\t"
               "global_load_async_to_lds_b128 %0, %1, %2 offset:16"
               :: "v"(ldsDst), "v"(gOff), "s"(sBase) : "memory");
}
__device__ __forceinline__ void waitAsync0() {
  asm volatile("s_wait_asynccnt 0x0" ::: "memory");
}

__global__ void kInit(unsigned* gMaxPos, unsigned* gMaxNeg, unsigned* gHasPos) {
  int i = threadIdx.x;
  gMaxPos[i] = ENC_NEG_INF;
  gMaxNeg[i] = ENC_NEG_INF;
  gHasPos[i] = 0u;
}

__global__ void kThr(const unsigned* gMaxPos, const unsigned* gMaxNeg,
                     float* posThr, float* negThr) {
  int i = threadIdx.x;
  posThr[i] = decf(gMaxNeg[i]) + MARGIN_F;                    // max_neg + margin
  negThr[i] = fmaxf(NEGBASE_F, decf(gMaxPos[i])) - MARGIN_F;  // max(0.6,max_pos)-margin
}

// B (65536x512) f32 -> bf16, row-major. 8 floats per thread.
__global__ __launch_bounds__(256) void kConvB(const float* __restrict__ B,
                                              unsigned* __restrict__ B16) {
  const size_t t = (size_t)blockIdx.x * 256 + threadIdx.x;
  const float4* src = (const float4*)B + t * 2;
  float4 a = src[0], b = src[1];
  uint4 q;
  q.x = pk_bf16(a.x, a.y); q.y = pk_bf16(a.z, a.w);
  q.z = pk_bf16(b.x, b.y); q.w = pk_bf16(b.z, b.w);
  ((uint4*)B16)[t] = q;
}

// A (512x512) f32 -> bf16 in WMMA A-fragment layout.
// Fragment block fb = rt*16 + ks: 32 lanes x 32B; lane L (m=L&15, half=L>>4)
// holds row rt*16+m, K {k0+16h..+15} contiguous.
__global__ __launch_bounds__(32) void kPackA(const float* __restrict__ A,
                                             unsigned* __restrict__ aPack) {
  const int fb = blockIdx.x;          // 0..511
  const int rt = fb >> 4, ks = fb & 15;
  const int lane = threadIdx.x;
  const int m = lane & 15, half = lane >> 4;
  const float* aRow = A + (size_t)(rt * 16 + m) * K_TOT;
  const int k0 = ks * 32;
  const int kA = k0 + 16 * half;
  const int kB = k0 + 16 * half + 8;
  float4 f0 = *(const float4*)(aRow + kA);
  float4 f1 = *(const float4*)(aRow + kA + 4);
  float4 f2 = *(const float4*)(aRow + kB);
  float4 f3 = *(const float4*)(aRow + kB + 4);
  uint4 qa0, qa1;
  qa0.x = pk_bf16(f0.x, f0.y); qa0.y = pk_bf16(f0.z, f0.w);
  qa0.z = pk_bf16(f1.x, f1.y); qa0.w = pk_bf16(f1.z, f1.w);
  qa1.x = pk_bf16(f2.x, f2.y); qa1.y = pk_bf16(f2.z, f2.w);
  qa1.z = pk_bf16(f3.x, f3.y); qa1.w = pk_bf16(f3.z, f3.w);
  uint4* dst = (uint4*)aPack + (size_t)fb * 64 + lane * 2;
  dst[0] = qa0; dst[1] = qa1;
}

// Block tile: 64 rows x 128 cols. 8 waves: (wr 0..3) x (wc 0..1).
// Each wave: 16 rows x 64 cols = 4 accumulators of 16x16.
template <int PASS, bool PRE>
__global__ __launch_bounds__(256) void kGemm(
    const unsigned* __restrict__ aPack,           // bf16 A fragments
    const int* __restrict__ tcol,
    const float* __restrict__ B,                  // f32 B (fallback path)
    const unsigned* __restrict__ B16,             // bf16 B (PRE path)
    const int* __restrict__ trow,
    unsigned* gMaxPos, unsigned* gMaxNeg, unsigned* gHasPos,
    const float* __restrict__ posThr, const float* __restrict__ negThr,
    float* __restrict__ partial) {
  // B tiles in LDS as bf16, column-major, 80B col stride (bank-conflict-free
  // b128 reads). Two buffers for async double-buffering.
  __shared__ unsigned ldsB[2 * (LDS_BUF_BYTES / 4)];
  __shared__ unsigned lMaxPos[64], lMaxNeg[64], lHas[64];
  __shared__ float    lLoss[64][2];

  const int tid  = threadIdx.x;
  const int lane = tid & 31;
  const int wave = tid >> 5;     // 0..7
  const int wr   = wave >> 1;    // 0..3 row-wave
  const int wc   = wave & 1;     // 0..1 col-wave
  const int m    = lane & 15;
  const int half = lane >> 4;

  const int rowBaseWG = blockIdx.x * 64;   // 8 row blocks (fast dim -> B reuse)
  const int colBaseWG = blockIdx.y * 128;  // 512 col blocks
  const int rowBase   = rowBaseWG + wr * 16;
  const int rt        = blockIdx.x * 4 + wr;

  if (PASS == 1 && tid < 64) { lMaxPos[tid] = 0u; lMaxNeg[tid] = 0u; lHas[tid] = 0u; }

  v8f acc[4] = {};

  // Per-wave A fragment stream (pre-packed bf16, fully coalesced 1KB/frag)
  const uint4* apA = (const uint4*)aPack + ((size_t)rt * 16) * 64 + lane * 2;

  // Cooperative B staging: thread t -> col lc = t>>1, k-half kh = (t&1)
  const int lc = tid >> 1;
  const int kh = tid & 1;
  const unsigned ldsDst0 =
      (unsigned)(uintptr_t)(&ldsB[0]) + (unsigned)(lc * 80 + kh * 32);
  const unsigned long long b16base = (unsigned long long)(uintptr_t)B16;
  const float* bRow = B + (size_t)(colBaseWG + lc) * K_TOT + kh * 16;

  // Per-wave B fragment read base (col group wc*64 + m, k-half by lane half)
  const char* rdBase =
      (const char*)ldsB + (unsigned)((wc * 64 + m) * 80 + half * 32);

  unsigned gOff  = (unsigned)(((colBaseWG + lc) * K_TOT + kh * 16) * 2);
  unsigned rdOff = 0;
  unsigned wrOff = LDS_BUF_BYTES;   // prologue fills buffer 0; loop fills 1,0,..

  if (PRE) {
    asyncPairB128(ldsDst0, gOff, b16base);  // tile ks=0 -> buffer 0
    gOff += 64u;
  }

#pragma unroll 1
  for (int ks = 0; ks < NKS; ++ks) {
    if (PRE) {
      waitAsync0();        // my chunks of current tile arrived
      __syncthreads();     // everyone's arrived; prior reads of other buf done
      if (ks + 1 < NKS) {  // DMA next tile while computing current
        asyncPairB128(ldsDst0 + wrOff, gOff, b16base);
        gOff += 64u;
        wrOff ^= LDS_BUF_BYTES;
      }
    } else {
      const float4* bp = (const float4*)(bRow + ks * 32);
      float4 g0 = bp[0], g1 = bp[1], g2 = bp[2], g3 = bp[3];
      uint4 s0, s1;
      s0.x = pk_bf16(g0.x, g0.y); s0.y = pk_bf16(g0.z, g0.w);
      s0.z = pk_bf16(g1.x, g1.y); s0.w = pk_bf16(g1.z, g1.w);
      s1.x = pk_bf16(g2.x, g2.y); s1.y = pk_bf16(g2.z, g2.w);
      s1.z = pk_bf16(g3.x, g3.y); s1.w = pk_bf16(g3.z, g3.w);
      __syncthreads();     // previous iteration's LDS reads done
      uint4* st = (uint4*)&ldsB[lc * 20 + kh * 8];
      st[0] = s0; st[1] = s1;
      __syncthreads();     // tile visible
    }

    // A fragment: two b128 from pre-packed stream
    union { v16bf v; uint4 q[2]; } ua;
    ua.q[0] = apA[0]; ua.q[1] = apA[1];
    apA += 64;             // next 1KB fragment

    // Preload all 4 B fragments, then fire 4 WMMAs back-to-back
    union { v16bf v; uint4 q[2]; } ub[4];
#pragma unroll
    for (int tt = 0; tt < 4; ++tt) {
      const uint4* pB = (const uint4*)(rdBase + rdOff + tt * 1280);
      ub[tt].q[0] = pB[0]; ub[tt].q[1] = pB[1];
    }
#pragma unroll
    for (int tt = 0; tt < 4; ++tt)
      acc[tt] = __builtin_amdgcn_wmma_f32_16x16x32_bf16(
          false, ua.v, false, ub[tt].v, (short)0, acc[tt], false, false);

    if (PRE) rdOff ^= LDS_BUF_BYTES;
  }

  // C/D layout: VGPR v, lanes 0-15 -> row v col lane; lanes 16-31 -> row v+8
  const int rBase = rowBase + half * 8;

  if (PASS == 1) {
    float mp[8], mn[8];
    unsigned hp = 0;
#pragma unroll
    for (int v = 0; v < 8; ++v) { mp[v] = -__builtin_inff(); mn[v] = -__builtin_inff(); }
#pragma unroll
    for (int tt = 0; tt < 4; ++tt) {
      const int j   = colBaseWG + wc * 64 + tt * 16 + m;
      const int trj = trow[j];
#pragma unroll
      for (int v = 0; v < 8; ++v) {
        const int i = rBase + v;
        const float s = acc[tt][v];
        const bool pos0 = (tcol[i] == trj);
        const bool pos  = pos0 ^ (i == j);   // pos_mask[:, :n] ^= eye
        if (pos)   { mp[v] = fmaxf(mp[v], s); hp |= (1u << v); }
        if (!pos0) { mn[v] = fmaxf(mn[v], s); }   // neg = ~pos0 (pre-xor)
      }
    }
#pragma unroll
    for (int off = 8; off >= 1; off >>= 1) {
#pragma unroll
      for (int v = 0; v < 8; ++v) {
        mp[v] = fmaxf(mp[v], __shfl_xor(mp[v], off, 16));
        mn[v] = fmaxf(mn[v], __shfl_xor(mn[v], off, 16));
      }
      hp |= __shfl_xor(hp, off, 16);
    }
    if (m == 0) {
#pragma unroll
      for (int v = 0; v < 8; ++v) {
        const int lr = wr * 16 + half * 8 + v;
        atomicMax(&lMaxPos[lr], encf(mp[v]));
        atomicMax(&lMaxNeg[lr], encf(mn[v]));
        if (hp & (1u << v)) atomicOr(&lHas[lr], 1u);
      }
    }
    __syncthreads();
    if (tid < 64) {
      atomicMax(&gMaxPos[rowBaseWG + tid], lMaxPos[tid]);
      atomicMax(&gMaxNeg[rowBaseWG + tid], lMaxNeg[tid]);
      if (lHas[tid]) atomicOr(&gHasPos[rowBaseWG + tid], 1u);
    }
  } else {
    float accR[8], pThr[8], nThr[8];
    int tc[8];
#pragma unroll
    for (int v = 0; v < 8; ++v) {
      const int i = rBase + v;
      accR[v] = 0.f; pThr[v] = posThr[i]; nThr[v] = negThr[i]; tc[v] = tcol[i];
    }
#pragma unroll
    for (int tt = 0; tt < 4; ++tt) {
      const int j   = colBaseWG + wc * 64 + tt * 16 + m;
      const int trj = trow[j];
#pragma unroll
      for (int v = 0; v < 8; ++v) {
        const float s = acc[tt][v];
        const bool pos0 = (tc[v] == trj);
        const bool pos  = pos0 ^ ((rBase + v) == j);
        float c = 0.f;
        if (pos && s < pThr[v])   c += 1.0f - s;  // hard positive
        if (!pos0 && s > nThr[v]) c += s;         // hard negative
        accR[v] += c;
      }
    }
#pragma unroll
    for (int off = 8; off >= 1; off >>= 1)
#pragma unroll
      for (int v = 0; v < 8; ++v) accR[v] += __shfl_xor(accR[v], off, 16);
    if (m == 0)
#pragma unroll
      for (int v = 0; v < 8; ++v) lLoss[wr * 16 + half * 8 + v][wc] = accR[v];
    __syncthreads();
    if (tid < 64)  // deterministic per-(row, colblock) partial, no float atomics
      partial[(size_t)(rowBaseWG + tid) * 512 + blockIdx.y] =
          lLoss[tid][0] + lLoss[tid][1];
  }
}

__global__ __launch_bounds__(64) void kRowReduce(const float* __restrict__ partial,
                                                 const unsigned* __restrict__ gHasPos,
                                                 float* __restrict__ rowSum) {
  __shared__ float red[64];
  const int r = blockIdx.x, t = threadIdx.x;
  const float* p = partial + (size_t)r * 512;
  float s = 0.f;
  for (int k = 0; k < 512; k += 64) s += p[k + t];
  red[t] = s; __syncthreads();
  for (int off = 32; off >= 1; off >>= 1) {
    if (t < off) red[t] += red[t + off];
    __syncthreads();
  }
  if (t == 0) rowSum[r] = gHasPos[r] ? red[0] : 0.f;  // where(has_pos, ., 0)
}

__global__ __launch_bounds__(512) void kFinal(const float* __restrict__ rowSum,
                                              float* __restrict__ out) {
  __shared__ float red[512];
  const int t = threadIdx.x;
  red[t] = rowSum[t]; __syncthreads();
  for (int off = 256; off >= 1; off >>= 1) {
    if (t < off) red[t] += red[t + off];
    __syncthreads();
  }
  if (t == 0) out[0] = red[0] * (1.0f / (float)R_TOT);
}

extern "C" void kernel_launch(void* const* d_in, const int* in_sizes, int n_in,
                              void* d_out, int out_size, void* d_ws, size_t ws_size,
                              hipStream_t stream) {
  const float* A    = (const float*)d_in[0];  // inputs_col  (512,512)
  const int*   tcol = (const int*)d_in[1];    // targets_col (512)
  const float* B    = (const float*)d_in[2];  // inputs_row  (65536,512)
  const int*   trow = (const int*)d_in[3];    // targets_row (65536)

  char* ws = (char*)d_ws;
  unsigned* gMaxPos = (unsigned*)(ws + 0);
  unsigned* gMaxNeg = (unsigned*)(ws + 2048);
  unsigned* gHasPos = (unsigned*)(ws + 4096);
  float*    posThr  = (float*)(ws + 6144);
  float*    negThr  = (float*)(ws + 8192);
  float*    rowSum  = (float*)(ws + 10240);
  float*    partial = (float*)(ws + 12288);            // 512*512 f32 = 1MB
  unsigned* aPack   = (unsigned*)(ws + 12288 + (1 << 20));      // 512KB
  unsigned* B16     = (unsigned*)(ws + (2 << 20));              // 64MB bf16 B
  float*    out     = (float*)d_out;

  const bool pre = ws_size >= ((size_t)(2 << 20) + ((size_t)C_TOT * K_TOT * 2));

  const dim3 grid(R_TOT / 64, C_TOT / 128);  // (8, 512)

  kInit<<<1, 512, 0, stream>>>(gMaxPos, gMaxNeg, gHasPos);
  kPackA<<<512, 32, 0, stream>>>(A, aPack);
  if (pre) {
    kConvB<<<(C_TOT * (K_TOT / 8)) / 256, 256, 0, stream>>>(B, B16);
    kGemm<1, true><<<grid, 256, 0, stream>>>(aPack, tcol, B, B16, trow, gMaxPos,
                                             gMaxNeg, gHasPos, posThr, negThr, partial);
    kThr<<<1, 512, 0, stream>>>(gMaxPos, gMaxNeg, posThr, negThr);
    kGemm<2, true><<<grid, 256, 0, stream>>>(aPack, tcol, B, B16, trow, gMaxPos,
                                             gMaxNeg, gHasPos, posThr, negThr, partial);
  } else {
    kGemm<1, false><<<grid, 256, 0, stream>>>(aPack, tcol, B, B16, trow, gMaxPos,
                                              gMaxNeg, gHasPos, posThr, negThr, partial);
    kThr<<<1, 512, 0, stream>>>(gMaxPos, gMaxNeg, posThr, negThr);
    kGemm<2, false><<<grid, 256, 0, stream>>>(aPack, tcol, B, B16, trow, gMaxPos,
                                              gMaxNeg, gHasPos, posThr, negThr, partial);
  }
  kRowReduce<<<512, 64, 0, stream>>>(partial, gHasPos, rowSum);
  kFinal<<<1, 512, 0, stream>>>(rowSum, out);
}
